// GraphPoolingClassifier_49813030699095
// MI455X (gfx1250) — compile-verified, matrix operand order checked by reading, not run on hardware
//
#include <hip/hip_runtime.h>
#include <hip/hip_bf16.h>

typedef __attribute__((ext_vector_type(16))) _Float16 v16h;
typedef __attribute__((ext_vector_type(4)))  _Float16 v4h;
typedef __attribute__((ext_vector_type(8)))  float    v8f;

#define NB     1024      // number of graphs (B)
#define DIM    512       // D == H
#define NCLS   52
#define BN_EPS 1e-5f

// ---------------------------------------------------------------------------
// 1) Exclusive prefix scan of per-graph node counts (1 block, 1024 threads)
// ---------------------------------------------------------------------------
__global__ void scan_kernel(const int* __restrict__ counts, int* __restrict__ off) {
    __shared__ int buf[2][NB];
    int t = threadIdx.x;
    buf[0][t] = counts[t];
    __syncthreads();
    int src = 0;
    for (int d = 1; d < NB; d <<= 1) {
        int v = buf[src][t];
        if (t >= d) v += buf[src][t - d];
        buf[src ^ 1][t] = v;
        __syncthreads();
        src ^= 1;
    }
    off[t] = (t == 0) ? 0 : buf[src][t - 1];   // exclusive scan
}

// ---------------------------------------------------------------------------
// 2) Segment-mean pooling: one block per graph, 128 threads x 4 consecutive
//    columns (float4 / global_load_b128 per row). Two independent accumulators
//    break the FMA dependency chain so >=2 b128 loads stay in flight per lane.
//    Single pass over the dominant 419 MB of HBM traffic. f16 output.
// ---------------------------------------------------------------------------
__global__ void pool_kernel(const float* __restrict__ z,
                            const int* __restrict__ off,
                            const int* __restrict__ counts,
                            _Float16* __restrict__ feats) {
    int g = blockIdx.x;
    int t = threadIdx.x;                       // 0..127, owns cols 4t..4t+3
    int start = off[g];
    int cnt   = counts[g];
    const float4* p = (const float4*)(z + (size_t)start * DIM) + t;

    float4 a0 = make_float4(0.f, 0.f, 0.f, 0.f);
    float4 a1 = make_float4(0.f, 0.f, 0.f, 0.f);
    int r = 0;
    for (; r + 1 < cnt; r += 2) {
        float4 v0 = p[(size_t)r * 128];
        float4 v1 = p[(size_t)(r + 1) * 128];
        a0.x += v0.x; a0.y += v0.y; a0.z += v0.z; a0.w += v0.w;
        a1.x += v1.x; a1.y += v1.y; a1.z += v1.z; a1.w += v1.w;
    }
    if (r < cnt) {
        float4 v0 = p[(size_t)r * 128];
        a0.x += v0.x; a0.y += v0.y; a0.z += v0.z; a0.w += v0.w;
    }
    float inv = 1.0f / (float)cnt;
    v4h o;
    o[0] = (_Float16)((a0.x + a1.x) * inv);
    o[1] = (_Float16)((a0.y + a1.y) * inv);
    o[2] = (_Float16)((a0.z + a1.z) * inv);
    o[3] = (_Float16)((a0.w + a1.w) * inv);
    *(v4h*)(feats + (size_t)g * DIM + 4 * t) = o;   // 8-byte store
}

// ---------------------------------------------------------------------------
// 3) Transpose f32 [K,N] weight -> f16 [N,K] (LDS-tiled), so WMMA B fragments
//    load as contiguous 32-byte vectors exactly like A fragments.
// ---------------------------------------------------------------------------
__global__ void transpose_cvt_kernel(const float* __restrict__ w,
                                     _Float16* __restrict__ wt) {
    __shared__ float tile[32][33];
    int k0 = blockIdx.x * 32, n0 = blockIdx.y * 32;
    int tx = threadIdx.x, ty = threadIdx.y;    // (32,32)
    tile[ty][tx] = w[(size_t)(k0 + ty) * DIM + n0 + tx];
    __syncthreads();
    wt[(size_t)(n0 + ty) * DIM + k0 + tx] = (_Float16)tile[tx][ty];
}

// ---------------------------------------------------------------------------
// 4) WMMA GEMM: C[M,N] = A[M,K](f16) * Bt[N,K](f16)^T + bias, f32 accumulate.
//    One wave per 16x16 tile; 16 x v_wmma_f32_16x16x32_f16 over K=512.
//    Fragment layout (C/D per CDNA5 ISA 7.12.2):
//      A/B (k order used here, same in A and B, not ISA table order): lane = row + 16*(k/16), halves h=0..15 <-> k%16  (contig 32B load)
//      C/D: c[i] <-> M = i + 8*(lane>=16), N = lane&15
// ---------------------------------------------------------------------------
__global__ void gemm_wmma_kernel(const _Float16* __restrict__ A,
                                 const _Float16* __restrict__ Bt,
                                 const float* __restrict__ bias,
                                 float* __restrict__ Cout,
                                 int M, int N, int K) {
    int wave = (int)((blockIdx.x * blockDim.x + threadIdx.x) >> 5);
    int lane = threadIdx.x & 31;
    int ntiles = N >> 4;
    int tm = wave / ntiles;
    int tn = wave % ntiles;
    int r  = lane & 15;
    int kg = lane >> 4;

    const _Float16* ap = A  + (size_t)(tm * 16 + r) * K + kg * 16;
    const _Float16* bp = Bt + (size_t)(tn * 16 + r) * K + kg * 16;

    v8f c = {};
    for (int ks = 0; ks < K; ks += 32) {
        v16h a = *(const v16h*)(ap + ks);
        v16h b = *(const v16h*)(bp + ks);
        c = __builtin_amdgcn_wmma_f32_16x16x32_f16(
                /*neg_a=*/false, a, /*neg_b=*/false, b,
                /*c_mod=*/(short)0, c, /*reuse_a=*/false, /*reuse_b=*/false);
    }

    int n  = tn * 16 + r;
    int m0 = tm * 16 + 8 * kg;
    float bv = bias[n];
    for (int i = 0; i < 8; ++i)
        Cout[(size_t)(m0 + i) * N + n] = c[i] + bv;
}

// ---------------------------------------------------------------------------
// 5) BatchNorm statistics: per-column sum and sum-of-squares
// ---------------------------------------------------------------------------
__global__ void zero_stats_kernel(float* __restrict__ s) {
    s[blockIdx.x * blockDim.x + threadIdx.x] = 0.f;
}

__global__ void bn_stats_kernel(const float* __restrict__ x,
                                float* __restrict__ sum,
                                float* __restrict__ sq) {
    int col = blockIdx.x * 256 + threadIdx.x;          // 2 blocks in x -> 512 cols
    int r0  = blockIdx.y * (NB / 16);                  // 16 row chunks of 64
    float s = 0.f, ss = 0.f;
    for (int r = r0; r < r0 + NB / 16; ++r) {
        float v = x[(size_t)r * DIM + col];
        s += v; ss += v * v;
    }
    atomicAdd(&sum[col], s);
    atomicAdd(&sq[col],  ss);
}

// ---------------------------------------------------------------------------
// 6) Fused BN (biased var) + ReLU + convert to f16
// ---------------------------------------------------------------------------
__global__ void bn_relu_cvt_kernel(const float* __restrict__ x,
                                   const float* __restrict__ sum,
                                   const float* __restrict__ sq,
                                   const float* __restrict__ gamma,
                                   const float* __restrict__ beta,
                                   _Float16* __restrict__ out) {
    int idx = blockIdx.x * 256 + threadIdx.x;          // NB*DIM elements
    int col = idx & (DIM - 1);
    const float invM = 1.0f / (float)NB;
    float m = sum[col] * invM;
    float v = sq[col] * invM - m * m;
    float y = gamma[col] * (x[idx] - m) * rsqrtf(v + BN_EPS) + beta[col];
    out[idx] = (_Float16)fmaxf(y, 0.f);
}

// ---------------------------------------------------------------------------
// 7) Final 512 -> 52 layer (27 MFLOP; plain f32 dot, all operands L2-resident)
// ---------------------------------------------------------------------------
__global__ void fc3_kernel(const _Float16* __restrict__ h,
                           const float* __restrict__ w3,
                           const float* __restrict__ b3,
                           float* __restrict__ out) {
    int idx = blockIdx.x * blockDim.x + threadIdx.x;
    if (idx >= NB * NCLS) return;
    int b = idx / NCLS, c = idx % NCLS;
    float acc = b3[c];
    const _Float16* hp = h + (size_t)b * DIM;
    for (int k = 0; k < DIM; ++k)
        acc += (float)hp[k] * w3[(size_t)k * NCLS + c];
    out[idx] = acc;
}

// ---------------------------------------------------------------------------
extern "C" void kernel_launch(void* const* d_in, const int* in_sizes, int n_in,
                              void* d_out, int out_size, void* d_ws, size_t ws_size,
                              hipStream_t stream) {
    const float* z      = (const float*)d_in[0];
    const int*   counts = (const int*)  d_in[1];
    const float* w1     = (const float*)d_in[2];
    const float* b1     = (const float*)d_in[3];
    const float* g1     = (const float*)d_in[4];
    const float* be1    = (const float*)d_in[5];
    const float* w2     = (const float*)d_in[6];
    const float* b2     = (const float*)d_in[7];
    const float* g2     = (const float*)d_in[8];
    const float* be2    = (const float*)d_in[9];
    const float* w3     = (const float*)d_in[10];
    const float* b3     = (const float*)d_in[11];
    float* out          = (float*)d_out;

    // ---- workspace layout (bytes) ----
    char* ws = (char*)d_ws;
    int*      off  = (int*)      (ws + 0);                 //  4 KB
    float*    sum  = (float*)    (ws + 4096);              //  2 KB
    float*    sq   = (float*)    (ws + 6144);              //  2 KB
    _Float16* wT   = (_Float16*) (ws + 8192);              // 512*512*2 = 512 KB (reused per layer)
    _Float16* Ah   = (_Float16*) (ws + 8192 + 524288);     // 1024*512*2 = 1 MB  (f16 activations, reused)
    float*    act  = (float*)    (ws + 8192 + 524288 + 1048576); // 1024*512*4 = 2 MB (f32 pre-BN)

    // 1) offsets + pooling -> Ah = f16 graph features
    scan_kernel<<<1, NB, 0, stream>>>(counts, off);
    pool_kernel<<<NB, 128, 0, stream>>>(z, off, counts, Ah);

    dim3 tb(32, 32), tg(DIM / 32, DIM / 32);
    const int gemmBlocks = (NB / 16) * (DIM / 16) * 32 / 256;  // 256 blocks of 8 waves

    // 2) layer 1: GEMM + BN + ReLU
    transpose_cvt_kernel<<<tg, tb, 0, stream>>>(w1, wT);
    gemm_wmma_kernel<<<gemmBlocks, 256, 0, stream>>>(Ah, wT, b1, act, NB, DIM, DIM);
    zero_stats_kernel<<<4, 256, 0, stream>>>(sum);              // zeroes sum(512)+sq(512)
    bn_stats_kernel<<<dim3(2, 16), 256, 0, stream>>>(act, sum, sq);
    bn_relu_cvt_kernel<<<NB * DIM / 256, 256, 0, stream>>>(act, sum, sq, g1, be1, Ah);

    // 3) layer 2: GEMM + BN + ReLU
    transpose_cvt_kernel<<<tg, tb, 0, stream>>>(w2, wT);
    gemm_wmma_kernel<<<gemmBlocks, 256, 0, stream>>>(Ah, wT, b2, act, NB, DIM, DIM);
    zero_stats_kernel<<<4, 256, 0, stream>>>(sum);
    bn_stats_kernel<<<dim3(2, 16), 256, 0, stream>>>(act, sum, sq);
    bn_relu_cvt_kernel<<<NB * DIM / 256, 256, 0, stream>>>(act, sum, sq, g2, be2, Ah);

    // 4) classifier head
    fc3_kernel<<<(NB * NCLS + 255) / 256, 256, 0, stream>>>(Ah, w3, b3, out);
}
